// SimpleGAT_45054206935303
// MI455X (gfx1250) — matrix operand order repaired and checked by reading, not compiled, not run on hardware
//
#include <hip/hip_runtime.h>
#include <hip/hip_bf16.h>

typedef unsigned short u16;
typedef unsigned int   u32;
typedef __attribute__((ext_vector_type(16))) __bf16 v16bf;
typedef __attribute__((ext_vector_type(8)))  float  v8f;

#define F_IN   128
#define HID    64
#define HEADS  4
#define H1DIM  (HEADS * HID)   // 256

// ---------- helpers ----------
__device__ __forceinline__ float lrelu(float x) { return x > 0.f ? x : 0.2f * x; }
__device__ __forceinline__ float elu1(float x)  { return x > 0.f ? x : (expf(x) - 1.f); }

// order-preserving float->uint encoding for atomicMax-based segment max
__device__ __forceinline__ u32 ford(float f) {
    u32 u = __float_as_uint(f);
    return (u & 0x80000000u) ? ~u : (u | 0x80000000u);
}
__device__ __forceinline__ float funord(u32 u) {
    return (u & 0x80000000u) ? __uint_as_float(u ^ 0x80000000u) : __uint_as_float(~u);
}
#define ORD_NEG_INF 0x007FFFFFu   // ford(-inf)

__device__ __forceinline__ u16 f2bf(float f) {  // round-to-nearest-even f32 -> bf16
    u32 u = __float_as_uint(f);
    return (u16)((u + 0x7FFFu + ((u >> 16) & 1u)) >> 16);
}

// ---------- utility kernels ----------
__global__ void k_cvt_bf16(const float* __restrict__ s, u16* __restrict__ d, int n) {
    int i = blockIdx.x * 256 + threadIdx.x;
    if (i < n) d[i] = f2bf(s[i]);
}
__global__ void k_fill_u32(u32* __restrict__ p, u32 v, int n) {
    int i = blockIdx.x * 256 + threadIdx.x;
    if (i < n) p[i] = v;
}

// Pack row-major f32 weight W[K][Nw] into WMMA-B register-order bf16:
//   Bp[(((nt*(K/32) + kc)*32 + lane)*16 + j]
//   lane<16 : col = nt*16+lane,      k = kc*32 + j       (j=0..15)
//   lane>=16: col = nt*16+(lane-16), k = kc*32 + 16 + j
// so each GEMM lane feeds B with two contiguous 16B loads.
__global__ void k_pack_b(const float* __restrict__ W, u16* __restrict__ Bp,
                         int K, int Nw) {
    int i = blockIdx.x * 256 + threadIdx.x;           // one thread per (nt,kc,lane)
    int total = (Nw / 16) * (K / 32) * 32;
    if (i >= total) return;
    int lane = i & 31;
    int kc   = (i >> 5) % (K / 32);
    int nt   = i / (32 * (K / 32));
    int col  = nt * 16 + (lane & 15);
    int kb   = kc * 32 + (lane >> 4) * 16;
    u16* dst = Bp + (size_t)i * 16;
#pragma unroll
    for (int j = 0; j < 16; ++j)
        dst[j] = f2bf(W[(size_t)(kb + j) * Nw + col]);
}

// ---------- bf16 WMMA GEMM: C[M,Nw] = A[M,K] * Bp ----------
// one wave computes a 16x(16*NT) strip; K fully unrolled (K/32 wmma per acc)
template<int K, int NT>
__global__ void k_gemm_bf16(const u16* __restrict__ A, const u16* __restrict__ Bp,
                            float* __restrict__ C, int Nw) {
    const int lane = threadIdx.x;            // 0..31 (wave32)
    const int tM   = blockIdx.y * 16;
    const int tN0  = blockIdx.x * (16 * NT);
    const int col  = lane & 15;
    const int half = lane >> 4;
    constexpr int KC = K / 32;

    v8f acc[NT] = {};
    const u16* arow  = A + (size_t)(tM + col) * K;
    const u16* bbase = Bp + ((size_t)blockIdx.x * NT * KC * 32) * 16;

#pragma unroll
    for (int kc = 0; kc < KC; ++kc) {
        union { v16bf v; uint4 q[2]; } a;
        // A k order used here (matches packed B): lanes<16 -> K = k0+[0..15]; lanes>=16 -> k0+[16..31]
        const u16* ap = arow + kc * 32 + half * 16;
        a.q[0] = *(const uint4*)(ap);
        a.q[1] = *(const uint4*)(ap + 8);
#pragma unroll
        for (int t = 0; t < NT; ++t) {
            union { v16bf v; uint4 q[2]; } b;
            const u16* bp = bbase + ((size_t)(t * KC + kc) * 32 + lane) * 16;
            b.q[0] = *(const uint4*)(bp);
            b.q[1] = *(const uint4*)(bp + 8);
            acc[t] = __builtin_amdgcn_wmma_f32_16x16x32_bf16(
                false, a.v, false, b.v, (short)0, acc[t], false, false);
        }
    }
    // C/D layout: VGPR r -> (M = half*8 + r, N = col)
#pragma unroll
    for (int t = 0; t < NT; ++t)
#pragma unroll
        for (int r = 0; r < 8; ++r)
            C[(size_t)(tM + half * 8 + r) * Nw + tN0 + t * 16 + col] = acc[t][r];
}

// ---------- attention logits ----------
__global__ void k_alpha1(const float* __restrict__ H, const float* __restrict__ asv,
                         const float* __restrict__ adv, float* __restrict__ As,
                         float* __restrict__ Ad, int n) {
    int i = blockIdx.x * 256 + threadIdx.x;
    if (i >= n * HEADS) return;
    int node = i >> 2, h = i & 3;
    const float* hp = H + (size_t)node * H1DIM + h * HID;
    const float* s  = asv + h * HID;
    const float* d  = adv + h * HID;
    float a = 0.f, b = 0.f;
    for (int c = 0; c < HID; ++c) { float v = hp[c]; a += v * s[c]; b += v * d[c]; }
    As[i] = a; Ad[i] = b;
}
__global__ void k_alpha2(const float* __restrict__ H, const float* __restrict__ asv,
                         const float* __restrict__ adv, float* __restrict__ As,
                         float* __restrict__ Ad, int n) {
    int i = blockIdx.x * 256 + threadIdx.x;
    if (i >= n) return;
    const float* hp = H + (size_t)i * HID;
    float a = 0.f, b = 0.f;
    for (int c = 0; c < HID; ++c) { float v = hp[c]; a += v * asv[c]; b += v * adv[c]; }
    As[i] = a; Ad[i] = b;
}

// ---------- edge passes: segment max / segment sum (softmax) ----------
template<int HH>
__global__ void k_edge_max(const int* __restrict__ src, const int* __restrict__ dst,
                           const float* __restrict__ As, const float* __restrict__ Ad,
                           u32* __restrict__ mx, int E, int ET) {
    int e = blockIdx.x * 256 + threadIdx.x;
    if (e >= ET) return;
    int s, d;
    if (e < E) { s = src[e]; d = dst[e]; } else { s = d = e - E; }
#pragma unroll
    for (int h = 0; h < HH; ++h) {
        float sc = lrelu(As[s * HH + h] + Ad[d * HH + h]);
        atomicMax(&mx[d * HH + h], ford(sc));
    }
}
template<int HH>
__global__ void k_edge_den(const int* __restrict__ src, const int* __restrict__ dst,
                           const float* __restrict__ As, const float* __restrict__ Ad,
                           const u32* __restrict__ mx, float* __restrict__ den,
                           int E, int ET) {
    int e = blockIdx.x * 256 + threadIdx.x;
    if (e >= ET) return;
    int s, d;
    if (e < E) { s = src[e]; d = dst[e]; } else { s = d = e - E; }
#pragma unroll
    for (int h = 0; h < HH; ++h) {
        float sc = lrelu(As[s * HH + h] + Ad[d * HH + h]);
        atomicAdd(&den[d * HH + h], expf(sc - funord(mx[d * HH + h])));
    }
}

// ---------- weighted scatter-add (one block per edge) ----------
__global__ void k_scatter1(const int* __restrict__ src, const int* __restrict__ dst,
                           const float* __restrict__ As, const float* __restrict__ Ad,
                           const u32* __restrict__ mx, const float* __restrict__ den,
                           const float* __restrict__ H, float* __restrict__ O, int E) {
    __shared__ float coef[HEADS];
    __shared__ int sn, dn;
    int e = blockIdx.x, t = threadIdx.x;       // block = 256 = HEADS*HID
    if (t == 0) {
        int s, d;
        if (e < E) { s = src[e]; d = dst[e]; } else { s = d = e - E; }
        sn = s; dn = d;
    }
    __syncthreads();
    if (t < HEADS) {
        float sc = lrelu(As[sn * HEADS + t] + Ad[dn * HEADS + t]);
        float ex = expf(sc - funord(mx[dn * HEADS + t]));
        coef[t] = ex / (den[dn * HEADS + t] + 1e-16f);
    }
    __syncthreads();
    float v = H[(size_t)sn * H1DIM + t] * coef[t >> 6];
    atomicAdd(&O[(size_t)dn * H1DIM + t], v);
}
__global__ void k_scatter2(const int* __restrict__ src, const int* __restrict__ dst,
                           const float* __restrict__ As, const float* __restrict__ Ad,
                           const u32* __restrict__ mx, const float* __restrict__ den,
                           const float* __restrict__ H, float* __restrict__ O, int E) {
    __shared__ float coef;
    __shared__ int sn, dn;
    int e = blockIdx.x, t = threadIdx.x;       // block = 64 = HID
    if (t == 0) {
        int s, d;
        if (e < E) { s = src[e]; d = dst[e]; } else { s = d = e - E; }
        sn = s; dn = d;
        float sc = lrelu(As[s] + Ad[d]);
        float ex = expf(sc - funord(mx[d]));
        coef = ex / (den[d] + 1e-16f);
    }
    __syncthreads();
    float v = H[(size_t)sn * HID + t] * coef;
    atomicAdd(&O[(size_t)dn * HID + t], v);
}

// ---------- activations ----------
__global__ void k_act1(const float* __restrict__ O, const float* __restrict__ b,
                       u16* __restrict__ Hb, int n) {
    int i = blockIdx.x * 256 + threadIdx.x;
    if (i >= n) return;
    Hb[i] = f2bf(elu1(O[i] + b[i & (H1DIM - 1)]));
}
__global__ void k_act2(float* __restrict__ O, const float* __restrict__ b, int n) {
    int i = blockIdx.x * 256 + threadIdx.x;
    if (i >= n) return;
    O[i] = elu1(O[i] + b[i & (HID - 1)]);
}

// ---------- readout ----------
__global__ void k_pool(const float* __restrict__ H, float* __restrict__ pooled, int n) {
    __shared__ float red[256];
    int c = blockIdx.x;                          // 0..HID-1
    float acc = 0.f;
    for (int i = threadIdx.x; i < n; i += 256) acc += H[(size_t)i * HID + c];
    red[threadIdx.x] = acc;
    __syncthreads();
    for (int s = 128; s > 0; s >>= 1) {
        if (threadIdx.x < s) red[threadIdx.x] += red[threadIdx.x + s];
        __syncthreads();
    }
    if (threadIdx.x == 0) pooled[c] = red[0] / (float)n;
}
__global__ void k_final(const float* __restrict__ pooled, const float* __restrict__ Wc,
                        const float* __restrict__ bc, float* __restrict__ out) {
    int k = threadIdx.x;
    if (k < 2) {
        float acc = bc[k];
        for (int c = 0; c < HID; ++c) acc += pooled[c] * Wc[c * 2 + k];
        out[k] = acc;
    }
}

// ---------- host-side launch ----------
extern "C" void kernel_launch(void* const* d_in, const int* in_sizes, int n_in,
                              void* d_out, int out_size, void* d_ws, size_t ws_size,
                              hipStream_t stream) {
    const float* x    = (const float*)d_in[0];
    const int*   ei   = (const int*)  d_in[1];
    const float* W1   = (const float*)d_in[2];
    const float* as1w = (const float*)d_in[3];
    const float* ad1w = (const float*)d_in[4];
    const float* b1   = (const float*)d_in[5];
    const float* W2   = (const float*)d_in[6];
    const float* as2w = (const float*)d_in[7];
    const float* ad2w = (const float*)d_in[8];
    const float* b2   = (const float*)d_in[9];
    const float* Wc   = (const float*)d_in[10];
    const float* bc   = (const float*)d_in[11];
    float* out = (float*)d_out;

    const int N  = in_sizes[0] / F_IN;   // 50000 (divisible by 16)
    const int E  = in_sizes[1] / 2;      // 1,600,000
    const int ET = E + N;                // + self loops

    char* ws = (char*)d_ws;
    size_t off = 0;
    auto alloc = [&](size_t bytes) -> char* {
        char* p = ws + off;
        off += (bytes + 255) & ~(size_t)255;
        return p;
    };
    u16*   xb  = (u16*)  alloc((size_t)N * F_IN * 2);
    u16*   w1p = (u16*)  alloc((size_t)F_IN * H1DIM * 2);   // packed B layout
    u16*   w2p = (u16*)  alloc((size_t)H1DIM * HID * 2);    // packed B layout
    float* H1  = (float*)alloc((size_t)N * H1DIM * 4);
    float* O1  = (float*)alloc((size_t)N * H1DIM * 4);
    float* As1 = (float*)alloc((size_t)N * HEADS * 4);
    float* Ad1 = (float*)alloc((size_t)N * HEADS * 4);
    u32*   Mx1 = (u32*)  alloc((size_t)N * HEADS * 4);
    float* Dn1 = (float*)alloc((size_t)N * HEADS * 4);
    float* H2  = (float*)alloc((size_t)N * HID * 4);
    float* O2  = (float*)alloc((size_t)N * HID * 4);
    float* As2 = (float*)alloc((size_t)N * 4);
    float* Ad2 = (float*)alloc((size_t)N * 4);
    u32*   Mx2 = (u32*)  alloc((size_t)N * 4);
    float* Dn2 = (float*)alloc((size_t)N * 4);
    float* pooled = (float*)alloc(HID * 4);
    u16*   H1b = (u16*)H1;   // reuse H1 region for bf16 activations (consumed before write)

    const int* srcA = ei;
    const int* dstA = ei + E;
    auto cdiv = [](int a, int b) { return (a + b - 1) / b; };

    // operand prep: A -> bf16 row-major, W -> WMMA-B packed bf16
    k_cvt_bf16<<<cdiv(N * F_IN, 256), 256, 0, stream>>>(x, xb, N * F_IN);
    k_pack_b<<<cdiv((H1DIM / 16) * (F_IN / 32) * 32, 256), 256, 0, stream>>>(W1, w1p, F_IN, H1DIM);
    k_pack_b<<<cdiv((HID / 16) * (H1DIM / 32) * 32, 256), 256, 0, stream>>>(W2, w2p, H1DIM, HID);

    // per-call init (deterministic)
    k_fill_u32<<<cdiv(N * HEADS, 256), 256, 0, stream>>>(Mx1, ORD_NEG_INF, N * HEADS);
    k_fill_u32<<<cdiv(N * HEADS, 256), 256, 0, stream>>>((u32*)Dn1, 0u, N * HEADS);
    k_fill_u32<<<cdiv(N * H1DIM, 256), 256, 0, stream>>>((u32*)O1, 0u, N * H1DIM);
    k_fill_u32<<<cdiv(N, 256), 256, 0, stream>>>(Mx2, ORD_NEG_INF, N);
    k_fill_u32<<<cdiv(N, 256), 256, 0, stream>>>((u32*)Dn2, 0u, N);
    k_fill_u32<<<cdiv(N * HID, 256), 256, 0, stream>>>((u32*)O2, 0u, N * HID);

    // ---- layer 1 ----  C[50000,256] = A[50000,128] x B[128,256]
    k_gemm_bf16<F_IN, 4><<<dim3(H1DIM / 64, N / 16), 32, 0, stream>>>(xb, w1p, H1, H1DIM);
    k_alpha1<<<cdiv(N * HEADS, 256), 256, 0, stream>>>(H1, as1w, ad1w, As1, Ad1, N);
    k_edge_max<HEADS><<<cdiv(ET, 256), 256, 0, stream>>>(srcA, dstA, As1, Ad1, Mx1, E, ET);
    k_edge_den<HEADS><<<cdiv(ET, 256), 256, 0, stream>>>(srcA, dstA, As1, Ad1, Mx1, Dn1, E, ET);
    k_scatter1<<<ET, 256, 0, stream>>>(srcA, dstA, As1, Ad1, Mx1, Dn1, H1, O1, E);
    k_act1<<<cdiv(N * H1DIM, 256), 256, 0, stream>>>(O1, b1, H1b, N * H1DIM);

    // ---- layer 2 ----  C[50000,64] = A[50000,256] x B[256,64]
    k_gemm_bf16<H1DIM, 4><<<dim3(HID / 64, N / 16), 32, 0, stream>>>(H1b, w2p, H2, HID);
    k_alpha2<<<cdiv(N, 256), 256, 0, stream>>>(H2, as2w, ad2w, As2, Ad2, N);
    k_edge_max<1><<<cdiv(ET, 256), 256, 0, stream>>>(srcA, dstA, As2, Ad2, Mx2, E, ET);
    k_edge_den<1><<<cdiv(ET, 256), 256, 0, stream>>>(srcA, dstA, As2, Ad2, Mx2, Dn2, E, ET);
    k_scatter2<<<ET, 64, 0, stream>>>(srcA, dstA, As2, Ad2, Mx2, Dn2, H2, O2, E);
    k_act2<<<cdiv(N * HID, 256), 256, 0, stream>>>(O2, b2, N * HID);

    // ---- readout ----
    k_pool<<<HID, 256, 0, stream>>>(O2, pooled, N);
    k_final<<<1, 32, 0, stream>>>(pooled, Wc, bc, out);
}
